// HyperAttentionDTI_40029095198823
// MI455X (gfx1250) — matrix operand order repaired and checked by reading, not compiled, not run on hardware
//
#include <hip/hip_runtime.h>
#include <hip/hip_bf16.h>

// ---------------------------------------------------------------------------
// HyperAttentionDTI inference for MI455X (gfx1250, wave32, WMMA).
// All matrix math via v_wmma_f32_16x16x32_bf16 (bf16 in, f32 accumulate).
// Key algebraic optimization: the attW linear commutes with the mean over
// tokens, shrinking the 47-GFLOP broadcast einsum to 150M VALU ops + two
// small WMMA GEMMs.
// ---------------------------------------------------------------------------

typedef __attribute__((ext_vector_type(16))) __bf16 v16bf;
typedef __attribute__((ext_vector_type(8)))  float  v8f;

#define ACT_NONE  0
#define ACT_RELU  1
#define ACT_SIG   2
#define ACT_LEAKY 3

__device__ __forceinline__ unsigned short f2bf(float f) {
  unsigned u = __float_as_uint(f);
  unsigned r = u + 0x7fffu + ((u >> 16) & 1u);   // round-to-nearest-even
  return (unsigned short)(r >> 16);
}

union FragU { v16bf v; unsigned u[8]; };

// ---------------------------------------------------------------------------
// Generic implicit-GEMM / conv1d kernel, one wave per 16x16 output tile.
//   out[b][l][n] = act( sum_{kw,ci} A[b][l+kw][ci] * W[n][kw*CIN+ci] + bias[n] )
// A: bf16 token-major [16][LIN][CIN]; W: bf16 [CoutPad][KPAD] (zero padded).
// M = 16*LOUT (always multiple of 16 since B==16).
// ---------------------------------------------------------------------------
template<int CIN,int KW,int COUT,int LIN,int LOUT,int KPAD,int ACT,bool OF32,bool OBF>
__global__ __launch_bounds__(256) void gemm_wmma(
    const unsigned short* __restrict__ A,
    const unsigned short* __restrict__ W,
    const float*          __restrict__ bias,
    unsigned short*       __restrict__ obf,
    float*                __restrict__ of32)
{
  constexpr int NT   = (COUT + 15) / 16;
  constexpr int MT   = LOUT;          // M = 16*LOUT, B = 16
  constexpr int KTOT = KW * CIN;

  const int wave = blockIdx.x * 8 + (threadIdx.x >> 5);
  if (wave >= MT * NT) return;        // wave-uniform exit (EXEC stays full)
  const int lane = threadIdx.x & 31;
  const int h    = lane >> 4;         // lane half (K split per ISA layouts)
  const int ln   = lane & 15;
  const int mt   = wave / NT;
  const int nt   = wave % NT;

  const int row = mt * 16 + ln;       // A row for this lane (m = lane&15)
  const int b   = row / LOUT;
  const int l   = row - b * LOUT;
  const unsigned short* Abase = A + (b * LIN + l) * CIN;
  const int n = nt * 16 + ln;         // B/D column for this lane
  const unsigned short* Wbase = W + n * KPAD;

  v8f acc = {0.f,0.f,0.f,0.f,0.f,0.f,0.f,0.f};

  for (int k0 = 0; k0 < KPAD; k0 += 32) {
    FragU fa, fb;
#pragma unroll
    for (int j = 0; j < 8; ++j) {
      // 16-bit A layout: pair j covers k, k+1 with k = 16*(j>>2) + 8h + 2*(j&3)
      int kk = k0 + 16 * (j >> 2) + 8 * h + 2 * (j & 3);
      unsigned val = 0u;
      if (KTOT == KPAD || kk < KTOT) {
        int kw = kk / CIN;            // CIN is constexpr -> cheap
        int ci = kk - kw * CIN;       // CIN even -> pair stays contiguous
        val = *(const unsigned*)(Abase + kw * CIN + ci);
      }
      fa.u[j] = val;
    }
#pragma unroll
    for (int j = 0; j < 8; ++j) {
      // B layout: same K order as A: pair j covers k = 16*(j>>2) + 8h + 2*(j&3)
      int kk = k0 + 16 * (j >> 2) + 8 * h + 2 * (j & 3);
      fb.u[j] = *(const unsigned*)(Wbase + kk);
    }
    acc = __builtin_amdgcn_wmma_f32_16x16x32_bf16(
        false, fa.v, false, fb.v, (short)0, acc, false, false);
  }

  if (n < COUT) {
    float bv = bias[n];
#pragma unroll
    for (int v = 0; v < 8; ++v) {
      int m  = v + 8 * h;             // C/D layout: vgpr v -> row v + 8*half
      int r  = mt * 16 + m;
      int ob = r / LOUT;
      int ol = r - ob * LOUT;
      float val = acc[v] + bv;
      if (ACT == ACT_RELU)  val = fmaxf(val, 0.f);
      if (ACT == ACT_SIG)   val = 1.f / (1.f + __expf(-val));
      if (ACT == ACT_LEAKY) val = (val > 0.f) ? val : 0.01f * val;
      int off = (ob * LOUT + ol) * COUT + n;
      if (OBF)  obf[off]  = f2bf(val);
      if (OF32) of32[off] = val;
    }
  }
}

// ---------------------------------------------------------------------------
// Weight prep: torch conv weight [Cout][Cin][Kw] -> bf16 [CoutPad][KPAD] with
// column index c = kw*Cin + ci, zero padding elsewhere.
// ---------------------------------------------------------------------------
__global__ void k_prep_convw(const float* __restrict__ w, unsigned short* __restrict__ out,
                             int Cout, int Cin, int Kw, int CoutP, int Kpad)
{
  int i = blockIdx.x * 256 + threadIdx.x;
  int total = CoutP * Kpad;
  if (i >= total) return;
  int co = i / Kpad, c = i - co * Kpad;
  unsigned short v = 0;
  if (co < Cout && c < Kw * Cin) {
    int k = c / Cin, ci = c - k * Cin;
    v = f2bf(w[(co * Cin + ci) * Kw + k]);
  }
  out[i] = v;
}

__global__ void k_f2bf(const float* __restrict__ in, unsigned short* __restrict__ out, int n)
{
  int i = blockIdx.x * 256 + threadIdx.x;
  if (i < n) out[i] = f2bf(in[i]);
}

// Embedding lookup -> token-major bf16 [16][L][64]
__global__ void k_embed(const int* __restrict__ tok, const float* __restrict__ emb,
                        unsigned short* __restrict__ out, int total)
{
  int i = blockIdx.x * 256 + threadIdx.x;
  if (i >= total) return;
  int ci = i & 63;
  int bl = i >> 6;
  out[i] = f2bf(emb[tok[bl] * 64 + ci]);
}

// S_d[b,d,o] = mean_p relu(d_att[b,d,o] + p_att[b,p,o])   (bf16 out)
__global__ void k_mean_over_p(const float* __restrict__ d_att, const float* __restrict__ p_att,
                              unsigned short* __restrict__ Sd_bf)
{
  int bd = blockIdx.x;                 // 16*49 blocks
  int b  = bd / 49;
  int o  = threadIdx.x;                // 160 threads
  float da = d_att[bd * 160 + o];
  const float* pp = p_att + b * 1179 * 160 + o;
  float s = 0.f;
  for (int p = 0; p < 1179; ++p) s += fmaxf(da + pp[p * 160], 0.f);
  Sd_bf[bd * 160 + o] = f2bf(s * (1.f / 1179.f));
}

// S_p[b,p,o] = mean_d relu(d_att[b,d,o] + p_att[b,p,o])   (bf16 out)
__global__ void k_mean_over_d(const float* __restrict__ d_att, const float* __restrict__ p_att,
                              unsigned short* __restrict__ Sp_bf)
{
  int bp = blockIdx.x;                 // 16*1179 blocks
  int b  = bp / 1179;
  int o  = threadIdx.x;                // 160 threads
  float pa = p_att[bp * 160 + o];
  const float* dd = d_att + b * 49 * 160 + o;
  float s = 0.f;
  for (int d = 0; d < 49; ++d) s += fmaxf(pa + dd[d * 160], 0.f);
  Sp_bf[bp * 160 + o] = f2bf(s * (1.f / 49.f));
}

// Fused attention gating + global max pool -> pair [16][320] (f32 + bf16)
__global__ void k_pool(const float* __restrict__ dc3, const float* __restrict__ comp,
                       const float* __restrict__ pc3, const float* __restrict__ prot,
                       float* __restrict__ pair, unsigned short* __restrict__ pair_bf)
{
  int b = blockIdx.x, t = threadIdx.x; // 16 blocks x 320 threads
  float m = -1e30f;
  if (t < 160) {
    for (int l = 0; l < 49; ++l) {
      int off = (b * 49 + l) * 160 + t;
      m = fmaxf(m, dc3[off] * (0.5f + comp[off]));
    }
  } else {
    int c = t - 160;
    for (int l = 0; l < 1179; ++l) {
      int off = (b * 1179 + l) * 160 + c;
      m = fmaxf(m, pc3[off] * (0.5f + prot[off]));
    }
  }
  pair[b * 320 + t]    = m;
  pair_bf[b * 320 + t] = f2bf(m);
}

// Final 512 -> 2 layer, f32 (too small for WMMA to matter)
__global__ void k_final(const float* __restrict__ h3, const float* __restrict__ outW,
                        const float* __restrict__ outb, float* __restrict__ out)
{
  int t = threadIdx.x;                 // 32 threads
  int b = t >> 1, j = t & 1;
  float s = outb[j];
  for (int c = 0; c < 512; ++c) s += h3[b * 512 + c] * outW[j * 512 + c];
  out[b * 2 + j] = s;
}

// ---------------------------------------------------------------------------

template<int CIN,int KW,int COUT,int LIN,int LOUT,int KPAD,int ACT,bool OF32,bool OBF>
static void launch_gemm(const unsigned short* A, const unsigned short* W, const float* bias,
                        unsigned short* obf, float* of32, hipStream_t s)
{
  constexpr int NT    = (COUT + 15) / 16;
  constexpr int tiles = LOUT * NT;
  int blocks = (tiles + 7) / 8;
  gemm_wmma<CIN,KW,COUT,LIN,LOUT,KPAD,ACT,OF32,OBF><<<blocks, 256, 0, s>>>(A, W, bias, obf, of32);
}

extern "C" void kernel_launch(void* const* d_in, const int* in_sizes, int n_in,
                              void* d_out, int out_size, void* d_ws, size_t ws_size,
                              hipStream_t stream)
{
  const int*   drug_tok = (const int*)  d_in[0];
  const int*   prot_tok = (const int*)  d_in[1];
  const float* drug_emb = (const float*)d_in[2];
  const float* prot_emb = (const float*)d_in[3];
  const float* dW1=(const float*)d_in[4],  *db1=(const float*)d_in[5];
  const float* dW2=(const float*)d_in[6],  *db2=(const float*)d_in[7];
  const float* dW3=(const float*)d_in[8],  *db3=(const float*)d_in[9];
  const float* pW1=(const float*)d_in[10], *pb1=(const float*)d_in[11];
  const float* pW2=(const float*)d_in[12], *pb2=(const float*)d_in[13];
  const float* pW3=(const float*)d_in[14], *pb3=(const float*)d_in[15];
  const float* dattW=(const float*)d_in[16], *dattb=(const float*)d_in[17];
  const float* pattW=(const float*)d_in[18], *pattb=(const float*)d_in[19];
  const float* attW =(const float*)d_in[20], *attb =(const float*)d_in[21];
  const float* fc1W=(const float*)d_in[22], *fc1b=(const float*)d_in[23];
  const float* fc2W=(const float*)d_in[24], *fc2b=(const float*)d_in[25];
  const float* fc3W=(const float*)d_in[26], *fc3b=(const float*)d_in[27];
  const float* outW=(const float*)d_in[28], *outb=(const float*)d_in[29];

  // Bump allocator over d_ws (256B aligned)
  char* base = (char*)d_ws;
  size_t off = 0;
  auto alloc = [&](size_t bytes) -> void* {
    void* r = base + off;
    off = (off + bytes + 255) & ~(size_t)255;
    return r;
  };

  // bf16 buffers
  unsigned short* de_bf   = (unsigned short*)alloc(65536u   * 2);
  unsigned short* pe_bf   = (unsigned short*)alloc(1228800u * 2);
  unsigned short* dc1_bf  = (unsigned short*)alloc(39040u   * 2);
  unsigned short* dc2_bf  = (unsigned short*)alloc(71680u   * 2);
  unsigned short* dc3_bf  = (unsigned short*)alloc(125440u  * 2);
  unsigned short* pc1_bf  = (unsigned short*)alloc(766080u  * 2);
  unsigned short* pc2_bf  = (unsigned short*)alloc(1523200u * 2);
  unsigned short* pc3_bf  = (unsigned short*)alloc(3018240u * 2);
  unsigned short* wtd1    = (unsigned short*)alloc(48u*256  * 2);
  unsigned short* wtd2    = (unsigned short*)alloc(80u*256  * 2);
  unsigned short* wtd3    = (unsigned short*)alloc(160u*640 * 2);
  unsigned short* wtp1    = (unsigned short*)alloc(48u*256  * 2);
  unsigned short* wtp2    = (unsigned short*)alloc(80u*320  * 2);
  unsigned short* wtp3    = (unsigned short*)alloc(160u*960 * 2);
  unsigned short* dattWb  = (unsigned short*)alloc(25600u   * 2);
  unsigned short* pattWb  = (unsigned short*)alloc(25600u   * 2);
  unsigned short* attWb   = (unsigned short*)alloc(25600u   * 2);
  unsigned short* fc1Wb   = (unsigned short*)alloc(327680u  * 2);
  unsigned short* fc2Wb   = (unsigned short*)alloc(1048576u * 2);
  unsigned short* fc3Wb   = (unsigned short*)alloc(524288u  * 2);
  unsigned short* Sd_bf   = (unsigned short*)alloc(125440u  * 2);
  unsigned short* Sp_bf   = (unsigned short*)alloc(3018240u * 2);
  unsigned short* pair_bf = (unsigned short*)alloc(5120u    * 2);
  unsigned short* h1_bf   = (unsigned short*)alloc(16384u   * 2);
  unsigned short* h2_bf   = (unsigned short*)alloc(16384u   * 2);
  // f32 buffers
  float* dc3_f  = (float*)alloc(125440u  * 4);
  float* pc3_f  = (float*)alloc(3018240u * 4);
  float* datt_f = (float*)alloc(125440u  * 4);   // reused for comp_atte
  float* patt_f = (float*)alloc(3018240u * 4);   // reused for prot_atte
  float* pair_f = (float*)alloc(5120u    * 4);
  float* h3_f   = (float*)alloc(8192u    * 4);
  (void)ws_size; (void)in_sizes; (void)n_in; (void)out_size;

  auto grid1 = [](int n) { return (n + 255) / 256; };

  // --- weight / embedding prep (deterministic each call) ---
  k_prep_convw<<<grid1(48*256),  256, 0, stream>>>(dW1, wtd1, 40, 64, 4,  48, 256);
  k_prep_convw<<<grid1(80*256),  256, 0, stream>>>(dW2, wtd2, 80, 40, 6,  80, 256);
  k_prep_convw<<<grid1(160*640), 256, 0, stream>>>(dW3, wtd3, 160,80, 8, 160, 640);
  k_prep_convw<<<grid1(48*256),  256, 0, stream>>>(pW1, wtp1, 40, 64, 4,  48, 256);
  k_prep_convw<<<grid1(80*320),  256, 0, stream>>>(pW2, wtp2, 80, 40, 8,  80, 320);
  k_prep_convw<<<grid1(160*960), 256, 0, stream>>>(pW3, wtp3, 160,80,12, 160, 960);
  k_f2bf<<<grid1(25600),   256, 0, stream>>>(dattW, dattWb, 25600);
  k_f2bf<<<grid1(25600),   256, 0, stream>>>(pattW, pattWb, 25600);
  k_f2bf<<<grid1(25600),   256, 0, stream>>>(attW,  attWb,  25600);
  k_f2bf<<<grid1(327680),  256, 0, stream>>>(fc1W,  fc1Wb,  327680);
  k_f2bf<<<grid1(1048576), 256, 0, stream>>>(fc2W,  fc2Wb,  1048576);
  k_f2bf<<<grid1(524288),  256, 0, stream>>>(fc3W,  fc3Wb,  524288);
  k_embed<<<grid1(65536),   256, 0, stream>>>(drug_tok, drug_emb, de_bf, 65536);
  k_embed<<<grid1(1228800), 256, 0, stream>>>(prot_tok, prot_emb, pe_bf, 1228800);

  // --- CNN stacks (implicit-GEMM WMMA) ---
  launch_gemm<64, 4, 40, 64,  61, 256, ACT_RELU, false, true >(de_bf,  wtd1, db1, dc1_bf, nullptr, stream);
  launch_gemm<40, 6, 80, 61,  56, 256, ACT_RELU, false, true >(dc1_bf, wtd2, db2, dc2_bf, nullptr, stream);
  launch_gemm<80, 8, 160,56,  49, 640, ACT_RELU, true,  true >(dc2_bf, wtd3, db3, dc3_bf, dc3_f,  stream);
  launch_gemm<64, 4, 40, 1200,1197,256, ACT_RELU, false, true >(pe_bf,  wtp1, pb1, pc1_bf, nullptr, stream);
  launch_gemm<40, 8, 80, 1197,1190,320, ACT_RELU, false, true >(pc1_bf, wtp2, pb2, pc2_bf, nullptr, stream);
  launch_gemm<80,12, 160,1190,1179,960, ACT_RELU, true,  true >(pc2_bf, wtp3, pb3, pc3_bf, pc3_f,  stream);

  // --- attention projections (WMMA GEMM, KW=1) ---
  launch_gemm<160,1,160,49,  49,  160, ACT_NONE, true, false>(dc3_bf, dattWb, dattb, nullptr, datt_f, stream);
  launch_gemm<160,1,160,1179,1179,160, ACT_NONE, true, false>(pc3_bf, pattWb, pattb, nullptr, patt_f, stream);

  // --- broadcast relu-sum means (mean commutes with attW linear) ---
  k_mean_over_p<<<16*49,   160, 0, stream>>>(datt_f, patt_f, Sd_bf);
  k_mean_over_d<<<16*1179, 160, 0, stream>>>(datt_f, patt_f, Sp_bf);

  // --- attW + sigmoid (reuse d_att / p_att f32 buffers for outputs) ---
  launch_gemm<160,1,160,49,  49,  160, ACT_SIG, true, false>(Sd_bf, attWb, attb, nullptr, datt_f, stream);
  launch_gemm<160,1,160,1179,1179,160, ACT_SIG, true, false>(Sp_bf, attWb, attb, nullptr, patt_f, stream);

  // --- gating + global max pool ---
  k_pool<<<16, 320, 0, stream>>>(dc3_f, datt_f, pc3_f, patt_f, pair_f, pair_bf);

  // --- MLP head (M=16 -> single M tile per GEMM) ---
  launch_gemm<320, 1,1024,1,1, 320,  ACT_LEAKY, false, true >(pair_bf, fc1Wb, fc1b, h1_bf, nullptr, stream);
  launch_gemm<1024,1,1024,1,1, 1024, ACT_LEAKY, false, true >(h1_bf,   fc2Wb, fc2b, h2_bf, nullptr, stream);
  launch_gemm<1024,1,512, 1,1, 1024, ACT_LEAKY, true,  false>(h2_bf,   fc3Wb, fc3b, nullptr, h3_f,  stream);

  k_final<<<1, 32, 0, stream>>>(h3_f, outW, outb, (float*)d_out);
}
